// MultiHeadAttention_51943334478079
// MI455X (gfx1250) — matrix operand order repaired and checked by reading, not compiled, not run on hardware
//
#include <hip/hip_runtime.h>

#define B_   2
#define S_   2048
#define H_   16
#define DK_  64
#define DM_  1024
#define M_   (B_*S_)      // 4096
#define LN_EPS 1e-5f

// LDS B-tile: 64 rows (N) x 32 (K) bf16, row padded 32->40 ushorts (80B) for banks
#define BROW_U16 40
#define BROW_U32 20

typedef __attribute__((ext_vector_type(16))) __bf16 bf16x16;
typedef __attribute__((ext_vector_type(8)))  float  f32x8;

union ABFrag { bf16x16 v; unsigned int u[8]; };
union CDFrag { f32x8 v; float f[8]; };

__device__ __forceinline__ unsigned short f2bf(float f) {
  unsigned int u = __float_as_uint(f);
  u += 0x7FFFu + ((u >> 16) & 1u);          // round-to-nearest-even
  return (unsigned short)(u >> 16);
}
__device__ __forceinline__ unsigned int pack2bf(float lo, float hi) {
  return (unsigned int)f2bf(lo) | ((unsigned int)f2bf(hi) << 16);
}

// CDNA5 async global->LDS copy (ASYNCcnt). GVS mode: uniform SGPR base +
// per-lane 32-bit byte offset; vdst VGPR holds the LDS byte address.
__device__ __forceinline__ void async_b128_to_lds(unsigned lds_addr,
                                                  const unsigned short* sbase,
                                                  unsigned byte_off) {
  asm volatile("global_load_async_to_lds_b128 %0, %1, %2"
               :: "v"(lds_addr), "v"(byte_off), "s"(sbase)
               : "memory");
}
__device__ __forceinline__ void wait_async0() {
  asm volatile("s_wait_asynccnt 0x0" ::: "memory");
}

// ---------------------------------------------------------------- conversions
__global__ void cvt_bf16_kernel(const float* __restrict__ src,
                                unsigned short* __restrict__ dst, int n) {
  int i = blockIdx.x * blockDim.x + threadIdx.x;
  if (i < n) dst[i] = f2bf(src[i]);
}

// src [rows, cols] fp32 row-major -> dst [cols, rows] bf16 (transpose)
__global__ void cvt_bf16_T_kernel(const float* __restrict__ src,
                                  unsigned short* __restrict__ dst,
                                  int rows, int cols) {
  int i = blockIdx.x * blockDim.x + threadIdx.x;
  if (i < rows * cols) {
    int r = i / cols, c = i % cols;
    dst[(size_t)c * rows + r] = f2bf(src[i]);
  }
}

// ------------------------------------------------- projection GEMM (bf16 WMMA)
// Block tile 128(M) x 64(N); 8 waves, wave = 16 rows x 64 cols.
// B tile (shared by all 8 waves) staged in LDS via double-buffered async DMA.
// WT is W^T: element B(k,n) = WT[n*DM_ + k].
// store_vt == 0 : dst [B,H,S,DK]   (Q, K heads)
// store_vt == 1 : dst [B,H,DK,S]   (V transposed for attn@V)
__global__ void proj_gemm_kernel(const unsigned short* __restrict__ Abf,
                                 const unsigned short* __restrict__ WT,
                                 unsigned short* __restrict__ dst,
                                 int store_vt) {
  __shared__ unsigned short Bs[2][64 * BROW_U16];
  const int tid  = threadIdx.x;
  const int lane = tid & 31, wave = tid >> 5;
  const int bn = blockIdx.x & 15;            // N tile (64 wide)
  const int bm = blockIdx.x >> 4;            // M block (128 rows)
  const int tm = bm * 8 + wave;
  const int l15 = lane & 15, hi = lane >> 4;

  // async-copy assignment: thread -> (row cn of B tile, 16B chunk cc)
  const int cn = tid >> 2, cc = tid & 3;
  const unsigned short* sbase = WT + (size_t)bn * 64 * DM_;      // uniform
  const unsigned voff0 = (unsigned)((cn * DM_ + cc * 8) * 2);    // bytes

  const unsigned int* A32 = (const unsigned int*)Abf;
  const int KW = DM_ / 2;
  const size_t arow = (size_t)(tm * 16 + l15) * KW;

  CDFrag acc[4];
#pragma unroll
  for (int s = 0; s < 4; ++s) acc[s].v = (f32x8){0,0,0,0,0,0,0,0};

  // prologue: stage 0 DMA + A fragment 0
  async_b128_to_lds((unsigned)(size_t)&Bs[0][cn * BROW_U16 + cc * 8], sbase, voff0);
  ABFrag a_cur;
#pragma unroll
  for (int v = 0; v < 8; ++v)
    a_cur.u[v] = A32[arow + ((v >> 2) << 3) + hi * 4 + (v & 3)];

  const int NK = DM_ / 32;                   // 32 K-steps
  for (int kt = 0; kt < NK; ++kt) {
    wait_async0();
    __syncthreads();                         // stage kt visible to all waves
    if (kt + 1 < NK)                         // overlap next DMA with compute
      async_b128_to_lds((unsigned)(size_t)&Bs[(kt + 1) & 1][cn * BROW_U16 + cc * 8],
                        sbase, voff0 + (unsigned)((kt + 1) * 64));
    ABFrag a_next = a_cur;
    if (kt + 1 < NK) {
#pragma unroll
      for (int v = 0; v < 8; ++v)
        a_next.u[v] = A32[arow + (kt + 1) * 16 + ((v >> 2) << 3) + hi * 4 + (v & 3)];
    }
    const unsigned int* BsU = (const unsigned int*)&Bs[kt & 1][0];
#pragma unroll
    for (int sub = 0; sub < 4; ++sub) {
      const int nn = sub * 16 + l15;
      ABFrag b;
#pragma unroll
      for (int v = 0; v < 8; ++v)
        b.u[v] = BsU[nn * BROW_U32 + ((v >> 2) << 3) + hi * 4 + (v & 3)];
      acc[sub].v = __builtin_amdgcn_wmma_f32_16x16x32_bf16(
          false, a_cur.v, false, b.v, (short)0, acc[sub].v, false, false);
    }
    a_cur = a_next;
  }

#pragma unroll
  for (int sub = 0; sub < 4; ++sub) {
    const int col = bn * 64 + sub * 16 + l15;
    const int h = col >> 6, d = col & 63;
#pragma unroll
    for (int r = 0; r < 8; ++r) {
      const int mrow = tm * 16 + hi * 8 + r;
      const int bb = mrow >> 11, ss = mrow & 2047;
      const size_t idx = store_vt
          ? ((size_t)((bb * H_ + h) * DK_ + d)) * S_ + ss
          : ((size_t)((bb * H_ + h) * S_ + ss)) * DK_ + d;
      dst[idx] = f2bf(acc[sub].f[r]);
    }
  }
}

// ------------------------------------------------------- scores = scale*Q.K^T
// K-loop is only 2 steps -> direct-global fragments (L2-resident operands).
__global__ void scores_gemm_kernel(const unsigned short* __restrict__ Qh,
                                   const unsigned short* __restrict__ Kh,
                                   const unsigned char*  __restrict__ mask,
                                   float* __restrict__ scores) {
  const int tid  = threadIdx.x;
  const int lane = tid & 31, wave = tid >> 5;
  const int bh = blockIdx.y;
  const int b  = bh / H_;
  const int waveTile = blockIdx.x * 8 + wave;  // 128 q-tiles x 32 k-tiles
  const int NT = S_ / 64;
  const int tq = waveTile / NT;
  const int tk = waveTile % NT;
  const int l15 = lane & 15, hi = lane >> 4;

  const unsigned int* Q32 = (const unsigned int*)(Qh + (size_t)bh * S_ * DK_);
  const unsigned int* K32 = (const unsigned int*)(Kh + (size_t)bh * S_ * DK_);
  const int KW = DK_ / 2;
  const int qrow = tq * 16 + l15;

  CDFrag acc[4];
#pragma unroll
  for (int s = 0; s < 4; ++s) acc[s].v = (f32x8){0,0,0,0,0,0,0,0};

#pragma unroll
  for (int kt = 0; kt < DK_ / 32; ++kt) {
    ABFrag a;
#pragma unroll
    for (int v = 0; v < 8; ++v)
      a.u[v] = Q32[(size_t)qrow * KW + kt * 16 + ((v >> 2) << 3) + hi * 4 + (v & 3)];
#pragma unroll
    for (int sub = 0; sub < 4; ++sub) {
      const int n = tk * 64 + sub * 16 + l15;
      ABFrag bfr;
#pragma unroll
      for (int v = 0; v < 8; ++v)
        bfr.u[v] = K32[(size_t)n * KW + kt * 16 + ((v >> 2) << 3) + hi * 4 + (v & 3)];
      acc[sub].v = __builtin_amdgcn_wmma_f32_16x16x32_bf16(
          false, a.v, false, bfr.v, (short)0, acc[sub].v, false, false);
    }
  }

  float* out = scores + (size_t)bh * S_ * S_;
  const unsigned char* mb = mask + (size_t)b * S_ * S_;
#pragma unroll
  for (int sub = 0; sub < 4; ++sub) {
    const int sk = tk * 64 + sub * 16 + l15;
#pragma unroll
    for (int r = 0; r < 8; ++r) {
      const int sq = tq * 16 + hi * 8 + r;
      float val = acc[sub].f[r] * 0.125f;      // 1/sqrt(64)
      if (mb[(size_t)sq * S_ + sk]) val = -1e9f;
      out[(size_t)sq * S_ + sk] = val;
    }
  }
}

// ---------------------------------------------------------- softmax (in place)
__global__ void softmax_kernel(float* __restrict__ attn) {
  __shared__ float red[256];
  const int tid = threadIdx.x;
  float* p = attn + (size_t)blockIdx.x * S_;

  float v[8], mx = -3.4e38f;
#pragma unroll
  for (int i = 0; i < 8; ++i) { v[i] = p[tid + i * 256]; mx = fmaxf(mx, v[i]); }
  red[tid] = mx; __syncthreads();
  for (int s = 128; s > 0; s >>= 1) {
    if (tid < s) red[tid] = fmaxf(red[tid], red[tid + s]);
    __syncthreads();
  }
  mx = red[0]; __syncthreads();

  float sum = 0.f;
#pragma unroll
  for (int i = 0; i < 8; ++i) { v[i] = __expf(v[i] - mx); sum += v[i]; }
  red[tid] = sum; __syncthreads();
  for (int s = 128; s > 0; s >>= 1) {
    if (tid < s) red[tid] += red[tid + s];
    __syncthreads();
  }
  const float inv = 1.f / red[0];
#pragma unroll
  for (int i = 0; i < 8; ++i) p[tid + i * 256] = v[i] * inv;
}

// --------------------------------------------------------- context = attn @ V
// All 8 waves of a block share the V^T head block -> async LDS staging of B.
// attn fp32 (cvt->bf16 on load), Vt [B,H,DK,S] bf16, ctx [B,S,H*DK] bf16.
__global__ void ctx_gemm_kernel(const float* __restrict__ attn,
                                const unsigned short* __restrict__ Vt,
                                unsigned short* __restrict__ ctx) {
  __shared__ unsigned short Bs[2][64 * BROW_U16];
  const int tid  = threadIdx.x;
  const int lane = tid & 31, wave = tid >> 5;
  const int bh = blockIdx.y;
  const int b  = bh / H_, h = bh % H_;
  const int tq = blockIdx.x * 8 + wave;        // 128 q-tiles per head
  const int l15 = lane & 15, hi = lane >> 4;

  const int cn = tid >> 2, cc = tid & 3;
  const unsigned short* sbase = Vt + (size_t)bh * DK_ * S_;      // uniform
  const unsigned voff0 = (unsigned)((cn * S_ + cc * 8) * 2);

  const float* arow = attn + (size_t)bh * S_ * S_ + (size_t)(tq * 16 + l15) * S_;

  CDFrag acc[4];
#pragma unroll
  for (int s = 0; s < 4; ++s) acc[s].v = (f32x8){0,0,0,0,0,0,0,0};

  async_b128_to_lds((unsigned)(size_t)&Bs[0][cn * BROW_U16 + cc * 8], sbase, voff0);
  ABFrag a_cur;
#pragma unroll
  for (int v = 0; v < 8; ++v) {
    const int k0 = ((v >> 2) << 4) + hi * 8 + (v & 3) * 2;
    const float2 f2 = *(const float2*)(arow + k0);
    a_cur.u[v] = pack2bf(f2.x, f2.y);
  }

  const int NK = S_ / 32;                      // 64 K-steps
  for (int kt = 0; kt < NK; ++kt) {
    wait_async0();
    __syncthreads();
    if (kt + 1 < NK)
      async_b128_to_lds((unsigned)(size_t)&Bs[(kt + 1) & 1][cn * BROW_U16 + cc * 8],
                        sbase, voff0 + (unsigned)((kt + 1) * 64));
    ABFrag a_next = a_cur;
    if (kt + 1 < NK) {
#pragma unroll
      for (int v = 0; v < 8; ++v) {
        const int k0 = (kt + 1) * 32 + ((v >> 2) << 4) + hi * 8 + (v & 3) * 2;
        const float2 f2 = *(const float2*)(arow + k0);
        a_next.u[v] = pack2bf(f2.x, f2.y);
      }
    }
    const unsigned int* BsU = (const unsigned int*)&Bs[kt & 1][0];
#pragma unroll
    for (int sub = 0; sub < 4; ++sub) {
      const int nn = sub * 16 + l15;
      ABFrag bfr;
#pragma unroll
      for (int v = 0; v < 8; ++v)
        bfr.u[v] = BsU[nn * BROW_U32 + ((v >> 2) << 3) + hi * 4 + (v & 3)];
      acc[sub].v = __builtin_amdgcn_wmma_f32_16x16x32_bf16(
          false, a_cur.v, false, bfr.v, (short)0, acc[sub].v, false, false);
    }
    a_cur = a_next;
  }

#pragma unroll
  for (int sub = 0; sub < 4; ++sub) {
    const int col = h * DK_ + sub * 16 + l15;
#pragma unroll
    for (int r = 0; r < 8; ++r) {
      const int s = tq * 16 + hi * 8 + r;
      ctx[((size_t)(b * S_ + s)) * DM_ + col] = f2bf(acc[sub].f[r]);
    }
  }
}

// --------------------------------------- output GEMM + residual (fp32 result)
__global__ void out_gemm_kernel(const unsigned short* __restrict__ ctx,
                                const unsigned short* __restrict__ WfcT,
                                const float* __restrict__ residual,
                                float* __restrict__ pre) {
  __shared__ unsigned short Bs[2][64 * BROW_U16];
  const int tid  = threadIdx.x;
  const int lane = tid & 31, wave = tid >> 5;
  const int bn = blockIdx.x & 15;
  const int bm = blockIdx.x >> 4;
  const int tm = bm * 8 + wave;
  const int l15 = lane & 15, hi = lane >> 4;

  const int cn = tid >> 2, cc = tid & 3;
  const unsigned short* sbase = WfcT + (size_t)bn * 64 * DM_;    // uniform
  const unsigned voff0 = (unsigned)((cn * DM_ + cc * 8) * 2);

  const unsigned int* A32 = (const unsigned int*)ctx;
  const int KW = DM_ / 2;
  const size_t arow = (size_t)(tm * 16 + l15) * KW;

  CDFrag acc[4];
#pragma unroll
  for (int s = 0; s < 4; ++s) acc[s].v = (f32x8){0,0,0,0,0,0,0,0};

  async_b128_to_lds((unsigned)(size_t)&Bs[0][cn * BROW_U16 + cc * 8], sbase, voff0);
  ABFrag a_cur;
#pragma unroll
  for (int v = 0; v < 8; ++v)
    a_cur.u[v] = A32[arow + ((v >> 2) << 3) + hi * 4 + (v & 3)];

  const int NK = DM_ / 32;
  for (int kt = 0; kt < NK; ++kt) {
    wait_async0();
    __syncthreads();
    if (kt + 1 < NK)
      async_b128_to_lds((unsigned)(size_t)&Bs[(kt + 1) & 1][cn * BROW_U16 + cc * 8],
                        sbase, voff0 + (unsigned)((kt + 1) * 64));
    ABFrag a_next = a_cur;
    if (kt + 1 < NK) {
#pragma unroll
      for (int v = 0; v < 8; ++v)
        a_next.u[v] = A32[arow + (kt + 1) * 16 + ((v >> 2) << 3) + hi * 4 + (v & 3)];
    }
    const unsigned int* BsU = (const unsigned int*)&Bs[kt & 1][0];
#pragma unroll
    for (int sub = 0; sub < 4; ++sub) {
      const int nn = sub * 16 + l15;
      ABFrag b;
#pragma unroll
      for (int v = 0; v < 8; ++v)
        b.u[v] = BsU[nn * BROW_U32 + ((v >> 2) << 3) + hi * 4 + (v & 3)];
      acc[sub].v = __builtin_amdgcn_wmma_f32_16x16x32_bf16(
          false, a_cur.v, false, b.v, (short)0, acc[sub].v, false, false);
    }
    a_cur = a_next;
  }

#pragma unroll
  for (int sub = 0; sub < 4; ++sub) {
    const int col = bn * 64 + sub * 16 + l15;
#pragma unroll
    for (int r = 0; r < 8; ++r) {
      const int mrow = tm * 16 + hi * 8 + r;
      const size_t idx = (size_t)mrow * DM_ + col;
      pre[idx] = acc[sub].f[r] + residual[idx];
    }
  }
}

// ------------------------------------------------------------------ LayerNorm
__global__ void layernorm_kernel(const float* __restrict__ pre,
                                 float* __restrict__ out) {
  __shared__ float red[256];
  const int tid = threadIdx.x;
  const float* p = pre + (size_t)blockIdx.x * DM_;

  float v[4], sum = 0.f;
#pragma unroll
  for (int i = 0; i < 4; ++i) { v[i] = p[tid + i * 256]; sum += v[i]; }
  red[tid] = sum; __syncthreads();
  for (int s = 128; s > 0; s >>= 1) {
    if (tid < s) red[tid] += red[tid + s];
    __syncthreads();
  }
  const float mu = red[0] * (1.0f / DM_); __syncthreads();

  float ss = 0.f;
#pragma unroll
  for (int i = 0; i < 4; ++i) { const float d = v[i] - mu; ss += d * d; }
  red[tid] = ss; __syncthreads();
  for (int s = 128; s > 0; s >>= 1) {
    if (tid < s) red[tid] += red[tid + s];
    __syncthreads();
  }
  const float inv = rsqrtf(red[0] * (1.0f / DM_) + LN_EPS);

  float* o = out + (size_t)blockIdx.x * DM_;
#pragma unroll
  for (int i = 0; i < 4; ++i) o[tid + i * 256] = (v[i] - mu) * inv;
}

// ------------------------------------------------------------------ launcher
extern "C" void kernel_launch(void* const* d_in, const int* in_sizes, int n_in,
                              void* d_out, int out_size, void* d_ws, size_t ws_size,
                              hipStream_t stream) {
  (void)in_sizes; (void)n_in; (void)out_size; (void)ws_size;

  const float* inQ  = (const float*)d_in[0];
  const float* inK  = (const float*)d_in[1];
  const float* inV  = (const float*)d_in[2];
  const unsigned char* mask = (const unsigned char*)d_in[3];
  const float* WQ   = (const float*)d_in[4];
  const float* WK   = (const float*)d_in[5];
  const float* WV   = (const float*)d_in[6];
  const float* Wfc  = (const float*)d_in[7];

  float* out_ln = (float*)d_out;                         // [B,S,DM] fp32
  float* attn   = (float*)d_out + (size_t)M_ * DM_;      // [B,H,S,S] fp32

  char* wsb = (char*)d_ws;
  size_t off = 0;
  auto carve = [&](size_t bytes) { void* p = wsb + off; off += bytes; return p; };

  unsigned short* Xq   = (unsigned short*)carve((size_t)M_ * DM_ * 2);
  unsigned short* Xk   = (unsigned short*)carve((size_t)M_ * DM_ * 2);
  unsigned short* Xv   = (unsigned short*)carve((size_t)M_ * DM_ * 2);
  unsigned short* WqT  = (unsigned short*)carve((size_t)DM_ * DM_ * 2);
  unsigned short* WkT  = (unsigned short*)carve((size_t)DM_ * DM_ * 2);
  unsigned short* WvT  = (unsigned short*)carve((size_t)DM_ * DM_ * 2);
  unsigned short* WfcT = (unsigned short*)carve((size_t)DM_ * DM_ * 2);
  unsigned short* Qh   = (unsigned short*)carve((size_t)M_ * DM_ * 2);  // [B,H,S,DK]
  unsigned short* Kh   = (unsigned short*)carve((size_t)M_ * DM_ * 2);  // [B,H,S,DK]
  unsigned short* Vt   = (unsigned short*)carve((size_t)M_ * DM_ * 2);  // [B,H,DK,S]
  unsigned short* ctx  = (unsigned short*)carve((size_t)M_ * DM_ * 2);  // [B,S,H*DK]
  float*          pre  = (float*)carve((size_t)M_ * DM_ * 4);

  const int nX = M_ * DM_;
  cvt_bf16_kernel<<<(nX + 255) / 256, 256, 0, stream>>>(inQ, Xq, nX);
  cvt_bf16_kernel<<<(nX + 255) / 256, 256, 0, stream>>>(inK, Xk, nX);
  cvt_bf16_kernel<<<(nX + 255) / 256, 256, 0, stream>>>(inV, Xv, nX);

  const int nW = DM_ * DM_;
  cvt_bf16_T_kernel<<<(nW + 255) / 256, 256, 0, stream>>>(WQ,  WqT,  DM_, DM_);
  cvt_bf16_T_kernel<<<(nW + 255) / 256, 256, 0, stream>>>(WK,  WkT,  DM_, DM_);
  cvt_bf16_T_kernel<<<(nW + 255) / 256, 256, 0, stream>>>(WV,  WvT,  DM_, DM_);
  cvt_bf16_T_kernel<<<(nW + 255) / 256, 256, 0, stream>>>(Wfc, WfcT, DM_, DM_);

  // projections: block = 128x64 tile, 512 blocks
  proj_gemm_kernel<<<512, 256, 0, stream>>>(Xq, WqT, Qh, 0);
  proj_gemm_kernel<<<512, 256, 0, stream>>>(Xk, WkT, Kh, 0);
  proj_gemm_kernel<<<512, 256, 0, stream>>>(Xv, WvT, Vt, 1);

  // scores + mask -> attn region of d_out; softmax in place
  scores_gemm_kernel<<<dim3(512, B_ * H_), 256, 0, stream>>>(Qh, Kh, mask, attn);
  softmax_kernel<<<B_ * H_ * S_, 256, 0, stream>>>(attn);

  // context = attn @ V ; output GEMM + residual ; LayerNorm
  ctx_gemm_kernel<<<dim3(16, B_ * H_), 256, 0, stream>>>(attn, Vt, ctx);
  out_gemm_kernel<<<512, 256, 0, stream>>>(ctx, WfcT, inQ, pre);
  layernorm_kernel<<<M_, 256, 0, stream>>>(pre, out_ln);
}
